// TrackNet_75239237091989
// MI455X (gfx1250) — matrix operand order repaired and checked by reading, not compiled, not run on hardware
//
#include <hip/hip_runtime.h>
#include <hip/hip_bf16.h>

typedef __attribute__((ext_vector_type(16))) _Float16 v16h;
typedef __attribute__((ext_vector_type(8)))  _Float16 v8h;
typedef __attribute__((ext_vector_type(8)))  float    v8f;

// ---------------------------------------------------------------------------
// Kernel 1: conf map (additive box scatter + sigmoid) fused with all three
// avg-pools. One block per (batch, 32x32 tile); the 224x224 map never
// touches HBM.
// ---------------------------------------------------------------------------
__global__ void confmap_pool_kernel(const float* __restrict__ confs,
                                    const float* __restrict__ boxes,
                                    float* __restrict__ att8,
                                    float* __restrict__ att16,
                                    float* __restrict__ att32) {
    const int b    = blockIdx.y;
    const int tile = blockIdx.x;          // 0..48 (7x7 tiles of 32x32)
    const int ty   = tile / 7, tx = tile % 7;
    const int tid  = threadIdx.x;         // 256 threads

    __shared__ float cf[32];
    __shared__ int   bx1[32], by1[32], bx2[32], by2[32];
    __shared__ float qs[256];
    __shared__ float s8s[16];
    __shared__ float s16s[4];

    if (tid < 32) {
        cf[tid] = confs[b * 32 + tid];
        const float* bp = boxes + (size_t)(b * 32 + tid) * 4;
        bx1[tid] = (int)floorf(bp[0] * 224.0f);
        by1[tid] = (int)floorf(bp[1] * 224.0f);
        bx2[tid] = (int)floorf(bp[2] * 224.0f);
        by2[tid] = (int)floorf(bp[3] * 224.0f);
    }
    __syncthreads();

    // each thread owns a 2x2 pixel quad of the 32x32 tile
    const int qy = tid >> 4, qx = tid & 15;
    const int Y0 = ty * 32 + qy * 2, X0 = tx * 32 + qx * 2;
    float qsum = 0.0f;
#pragma unroll
    for (int dy = 0; dy < 2; ++dy) {
#pragma unroll
        for (int dx = 0; dx < 2; ++dx) {
            const int y = Y0 + dy, x = X0 + dx;
            float s = 0.0f;
            for (int n = 0; n < 32; ++n) {
                bool in = (y >= by1[n]) & (y < by2[n]) & (x >= bx1[n]) & (x < bx2[n]);
                s += in ? cf[n] : 0.0f;
            }
            qsum += 1.0f / (1.0f + __expf(-s));   // sigmoid
        }
    }
    qs[tid] = qsum;
    __syncthreads();

    if (tid < 16) {                                // 4x4 att8 cells per tile
        const int ey = tid >> 2, ex = tid & 3;
        float s8 = 0.0f;
        for (int yy = 0; yy < 4; ++yy)
            for (int xx = 0; xx < 4; ++xx)
                s8 += qs[(ey * 4 + yy) * 16 + ex * 4 + xx];
        s8s[tid] = s8;
        att8[(size_t)b * 784 + (ty * 4 + ey) * 28 + (tx * 4 + ex)] = s8 * (1.0f / 64.0f);
    }
    __syncthreads();
    if (tid < 4) {                                 // 2x2 att16 cells per tile
        const int ey = tid >> 1, ex = tid & 1;
        float s16 = s8s[(2 * ey) * 4 + 2 * ex]     + s8s[(2 * ey) * 4 + 2 * ex + 1]
                  + s8s[(2 * ey + 1) * 4 + 2 * ex] + s8s[(2 * ey + 1) * 4 + 2 * ex + 1];
        s16s[tid] = s16;
        att16[(size_t)b * 196 + (ty * 2 + ey) * 14 + (tx * 2 + ex)] = s16 * (1.0f / 256.0f);
    }
    __syncthreads();
    if (tid == 0) {                                // 1 att32 cell per tile
        float s32 = s16s[0] + s16s[1] + s16s[2] + s16s[3];
        att32[(size_t)b * 49 + ty * 7 + tx] = s32 * (1.0f / 1024.0f);
    }
}

// ---------------------------------------------------------------------------
// Kernel 2: logits[b,c] = sum_hw feat[b,c,hw]*att[b,hw] via WMMA.
// Grid: (32 M-tiles of 16 channels, 16 batches); block = 4 waves, each wave
// K-strided over Kpad in steps of 32 (stride 128). Main body uses unguarded
// vector loads (wave-uniform tail condition); partial accumulators reduced
// through ds_add_f32. att is broadcast across all 16 N columns.
// ---------------------------------------------------------------------------
template <bool VEC4>
__global__ void logits_kernel(const float* __restrict__ feat,
                              const float* __restrict__ att,
                              float* __restrict__ logits,
                              int HW, int Kpad) {
    const int mt   = blockIdx.x;      // channel tile: c0 = mt*16
    const int b    = blockIdx.y;
    const int tid  = threadIdx.x;     // 128
    const int lane = tid & 31;
    const int wave = tid >> 5;        // 0..3
    const int hi   = lane >> 4;       // lane group 0/1
    const int mrow = lane & 15;       // M within 16-row tile

    __shared__ __align__(32) _Float16 att_s[800];  // Kpad <= 800
    __shared__ float lsum[16];

    const float* ab = att + (size_t)b * HW;
    for (int k = tid; k < Kpad; k += 128)
        att_s[k] = (k < HW) ? (_Float16)ab[k] : (_Float16)0.0f;
    if (tid < 16) lsum[tid] = 0.0f;
    __syncthreads();

    const int c0 = mt * 16;
    const float* frow = feat + (size_t)b * 512 * HW + (size_t)(c0 + mrow) * HW;

    v8f acc = {0.f, 0.f, 0.f, 0.f, 0.f, 0.f, 0.f, 0.f};
    for (int kb = wave * 32; kb < Kpad; kb += 128) {
        v16h a, bv;
        // B 32x16 f16 layout: lane needs K=8hi..8hi+7 and K=16+8hi..16+8hi+7
        // (two 8-half runs in LDS, padded, unguarded b128 loads).
        {
            const _Float16* bp = att_s + kb + 8 * hi;
            v8h b0 = *(const v8h*)bp;
            v8h b1 = *(const v8h*)(bp + 16);
#pragma unroll
            for (int i = 0; i < 8; ++i) { bv[i] = b0[i]; bv[8 + i] = b1[i]; }
        }
        // A 16x32 f16 layout: lane needs frow[kb+8hi .. +7] and
        // frow[kb+16+8hi .. +7] (two contiguous 8-float runs).
        if (kb + 32 <= HW) {               // wave-uniform: unguarded fast path
            const float* ap = frow + kb + 8 * hi;
            if (VEC4) {
                float4 f0 = *(const float4*)(ap);
                float4 f1 = *(const float4*)(ap + 4);
                float4 f2 = *(const float4*)(ap + 16);
                float4 f3 = *(const float4*)(ap + 20);
                a[0]  = (_Float16)f0.x; a[1]  = (_Float16)f0.y;
                a[2]  = (_Float16)f0.z; a[3]  = (_Float16)f0.w;
                a[4]  = (_Float16)f1.x; a[5]  = (_Float16)f1.y;
                a[6]  = (_Float16)f1.z; a[7]  = (_Float16)f1.w;
                a[8]  = (_Float16)f2.x; a[9]  = (_Float16)f2.y;
                a[10] = (_Float16)f2.z; a[11] = (_Float16)f2.w;
                a[12] = (_Float16)f3.x; a[13] = (_Float16)f3.y;
                a[14] = (_Float16)f3.z; a[15] = (_Float16)f3.w;
            } else {                       // rows not 16B-aligned (HW=49)
#pragma unroll
                for (int i = 0; i < 8; ++i) {
                    a[i]     = (_Float16)ap[i];
                    a[8 + i] = (_Float16)ap[16 + i];
                }
            }
        } else {                           // single tail iteration: guarded
#pragma unroll
            for (int i = 0; i < 8; ++i) {
                int k0 = kb + 8 * hi + i;
                int k1 = kb + 16 + 8 * hi + i;
                a[i]     = (k0 < HW) ? (_Float16)frow[k0] : (_Float16)0.0f;
                a[8 + i] = (k1 < HW) ? (_Float16)frow[k1] : (_Float16)0.0f;
            }
        }
        acc = __builtin_amdgcn_wmma_f32_16x16x32_f16(
            false, a, false, bv, (short)0, acc, false, false);
    }
    // D layout: VGPR v -> M = v (lanes0-15) / v+8 (lanes16-31); N replicated.
    if (mrow == 0) {
#pragma unroll
        for (int v = 0; v < 8; ++v)
            atomicAdd(&lsum[8 * hi + v], acc[v]);
    }
    __syncthreads();
    if (tid < 16)
        logits[b * 512 + c0 + tid] = lsum[tid];
}

// ---------------------------------------------------------------------------
// Kernel 3: softmax over C=512 per batch (tiny).
// ---------------------------------------------------------------------------
__global__ void softmax_kernel(const float* __restrict__ logits,
                               float* __restrict__ wout) {
    const int b   = blockIdx.x;
    const int tid = threadIdx.x;   // 256
    __shared__ float red_s[256];

    float l0 = logits[b * 512 + tid], l1 = logits[b * 512 + tid + 256];
    red_s[tid] = fmaxf(l0, l1);
    __syncthreads();
    for (int s = 128; s > 0; s >>= 1) {
        if (tid < s) red_s[tid] = fmaxf(red_s[tid], red_s[tid + s]);
        __syncthreads();
    }
    float mx = red_s[0];
    __syncthreads();
    float e0 = __expf(l0 - mx), e1 = __expf(l1 - mx);
    red_s[tid] = e0 + e1;
    __syncthreads();
    for (int s = 128; s > 0; s >>= 1) {
        if (tid < s) red_s[tid] += red_s[tid + s];
        __syncthreads();
    }
    float inv = 1.0f / red_s[0];
    wout[b * 512 + tid]       = e0 * inv;
    wout[b * 512 + tid + 256] = e1 * inv;
}

// ---------------------------------------------------------------------------
// Kernel 4: out[b,c,hw] = feat[b,c,hw] * w[b,c]; bandwidth-bound streaming.
// ---------------------------------------------------------------------------
__global__ void scale_vec4_kernel(const float* __restrict__ feat,
                                  const float* __restrict__ w,
                                  float* __restrict__ out,
                                  int HW, int total4, int lvl_off) {
    int i = blockIdx.x * blockDim.x + threadIdx.x;
    if (i >= total4) return;
    const int base = i * 4;
    const int CHW  = 512 * HW;
    const int b    = base / CHW;
    const int rem  = base - b * CHW;
    const int c    = rem / HW;                 // HW % 4 == 0 -> one c per vec4
    const float4 f = ((const float4*)feat)[i];
    const float  ww = w[b * 512 + c];
    float4 o; o.x = f.x * ww; o.y = f.y * ww; o.z = f.z * ww; o.w = f.w * ww;
    *(float4*)(out + (size_t)b * 526848 + lvl_off + rem) = o;
}

__global__ void scale_scalar_kernel(const float* __restrict__ feat,
                                    const float* __restrict__ w,
                                    float* __restrict__ out,
                                    int HW, int total, int lvl_off) {
    int i = blockIdx.x * blockDim.x + threadIdx.x;
    if (i >= total) return;
    const int CHW = 512 * HW;
    const int b   = i / CHW;
    const int rem = i - b * CHW;
    const int c   = rem / HW;
    out[(size_t)b * 526848 + lvl_off + rem] = feat[i] * w[b * 512 + c];
}

extern "C" void kernel_launch(void* const* d_in, const int* in_sizes, int n_in,
                              void* d_out, int out_size, void* d_ws, size_t ws_size,
                              hipStream_t stream) {
    (void)in_sizes; (void)n_in; (void)out_size; (void)ws_size;
    const float* confs = (const float*)d_in[0];
    const float* boxes = (const float*)d_in[1];
    const float* feat0 = (const float*)d_in[2];
    const float* feat1 = (const float*)d_in[3];
    const float* feat2 = (const float*)d_in[4];
    float* out = (float*)d_out;
    float* ws  = (float*)d_ws;

    // workspace layout (floats)
    float* att8  = ws;            // 16*28*28 = 12544
    float* att16 = ws + 12544;    // 16*14*14 =  3136
    float* att32 = ws + 15680;    // 16*7*7   =   784
    float* lg0   = ws + 16464;    // 16*512 each
    float* lg1   = lg0 + 8192;
    float* lg2   = lg1 + 8192;
    float* w0    = lg2 + 8192;
    float* w1    = w0 + 8192;
    float* w2    = w1 + 8192;

    // 1) conf map + pooled attention maps
    confmap_pool_kernel<<<dim3(49, 16), 256, 0, stream>>>(confs, boxes, att8, att16, att32);

    // 2) WMMA logits (512 blocks per level), then per-batch softmax
    logits_kernel<true ><<<dim3(32, 16), 128, 0, stream>>>(feat0, att8,  lg0, 784, 800);
    logits_kernel<true ><<<dim3(32, 16), 128, 0, stream>>>(feat1, att16, lg1, 196, 224);
    logits_kernel<false><<<dim3(32, 16), 128, 0, stream>>>(feat2, att32, lg2,  49,  64);
    softmax_kernel<<<16, 256, 0, stream>>>(lg0, w0);
    softmax_kernel<<<16, 256, 0, stream>>>(lg1, w1);
    softmax_kernel<<<16, 256, 0, stream>>>(lg2, w2);

    // 3) streaming rescale into concatenated output [B, 512*1029]
    {
        int total4 = 16 * 512 * 784 / 4;
        scale_vec4_kernel<<<(total4 + 255) / 256, 256, 0, stream>>>(feat0, w0, out, 784, total4, 0);
    }
    {
        int total4 = 16 * 512 * 196 / 4;
        scale_vec4_kernel<<<(total4 + 255) / 256, 256, 0, stream>>>(feat1, w1, out, 196, total4, 401408);
    }
    {
        int total = 16 * 512 * 49;
        scale_scalar_kernel<<<(total + 255) / 256, 256, 0, stream>>>(feat2, w2, out, 49, total, 501760);
    }
}
